// DGCNN_60610578481691
// MI455X (gfx1250) — matrix operand order repaired and checked by reading, not compiled, not run on hardware
//
#include <hip/hip_runtime.h>
#include <hip/hip_bf16.h>

// ---------------------------------------------------------------------------
// DGCNN forward for MI455X (gfx1250, wave32, WMMA).
//  - kNN distance GEMM: V_WMMA_F32_16X16X4_F32 (exact f32, ranking-safe),
//    A-fragments preloaded once per 16-row strip, B loads vectorized.
//  - edge-conv / fusion GEMMs: V_WMMA_F32_16X16X32_BF16 (bf16 in, f32 acc),
//    B-fragments are contiguous 32B runs -> single v16bf loads.
//  - all gathers go straight to global: every feature map fits in 192MB L2.
// ---------------------------------------------------------------------------

typedef __attribute__((ext_vector_type(2)))  float  v2f;
typedef __attribute__((ext_vector_type(8)))  float  v8f;
typedef __attribute__((ext_vector_type(8)))  __bf16 v8bf;
typedef __attribute__((ext_vector_type(16))) __bf16 v16bf;

#define DEV __device__ __forceinline__

DEV void atomicMaxF32Shared(float* addr, float val) {
  int* ai = (int*)addr;
  int cur = __float_as_int(*addr);
  while (__int_as_float(cur) < val) {
    int prev = atomicCAS(ai, cur, __float_as_int(val));
    if (prev == cur) break;
    cur = prev;
  }
}

// ------------------------------- helpers -----------------------------------

__global__ void sqnorm_kernel(const float* __restrict__ x, float* __restrict__ x2,
                              int total, int C) {
  int i = blockIdx.x * blockDim.x + threadIdx.x;
  if (i >= total) return;
  float s = 0.f;
  for (int c = 0; c < C; ++c) { float v = x[(size_t)i * C + c]; s += v * v; }
  x2[i] = s;
}

__global__ void f32_to_bf16_kernel(const float* __restrict__ a,
                                   __bf16* __restrict__ o, int n) {
  int i = blockIdx.x * blockDim.x + threadIdx.x;
  if (i < n) o[i] = (__bf16)a[i];
}

// alpha = g * rsqrt(v+eps); beta = (bias - m)*alpha + be  (folds conv bias + BN)
__global__ void alpha_beta_kernel(const float* __restrict__ g, const float* __restrict__ be,
                                  const float* __restrict__ m, const float* __restrict__ v,
                                  const float* __restrict__ bias,
                                  float* __restrict__ alpha, float* __restrict__ beta, int C) {
  int i = blockIdx.x * blockDim.x + threadIdx.x;
  if (i >= C) return;
  float al = g[i] * rsqrtf(v[i] + 1e-5f);
  alpha[i] = al;
  beta[i]  = (bias[i] - m[i]) * al + be[i];
}

__global__ void concat_bf16_kernel(const float* __restrict__ f0, const float* __restrict__ f1,
                                   const float* __restrict__ f2, const float* __restrict__ f3,
                                   __bf16* __restrict__ fcat, int BN) {
  int i = blockIdx.x * blockDim.x + threadIdx.x;
  if (i >= BN * 512) return;
  int bn = i >> 9, c = i & 511;
  float v;
  if      (c <  64) v = f0[(size_t)bn *  64 + c];
  else if (c < 128) v = f1[(size_t)bn *  64 + (c -  64)];
  else if (c < 256) v = f2[(size_t)bn * 128 + (c - 128)];
  else              v = f3[(size_t)bn * 256 + (c - 256)];
  fcat[i] = (__bf16)v;
}

// ------------------------------- kNN ---------------------------------------
// One wave per (batch, 16-row strip). C is a template constant so every
// bounds check folds; A-fragments (invariant over all 64 column tiles) are
// preloaded into VGPRs. dist = |xi|^2 + |xj|^2 - 2*dot (reference formula).
template<int C>
__global__ __launch_bounds__(32)
void knn_kernel(const float* __restrict__ x, const float* __restrict__ x2,
                int* __restrict__ idxout, int B, int N) {
  constexpr int KS = (C + 3) / 4;           // k-steps of 4
  __shared__ float dist[16][513];           // 513: conflict-free column reads
  __shared__ float x2s[512];
  int wg = blockIdx.x;
  int nt16 = N / 16;
  int b  = wg / nt16;
  int mt = wg % nt16;
  int lane = threadIdx.x;
  int lm = lane & 15;
  int hi = lane >> 4;                       // 0: K {0,1}, 1: K {2,3}
  int row = mt * 16 + lm;
  const float* xb  = x  + (size_t)b * N * C;
  const float* x2b = x2 + (size_t)b * N;
  const float* rowp = xb + (size_t)row * C;

  // Preload A fragments (v2f each, <=64 VGPRs at C=128).
  v2f afr[KS];
#pragma unroll
  for (int k = 0; k < KS; ++k) {
    if (C == 3) {                           // folds at compile time
      afr[k].x = hi ? rowp[2] : rowp[0];
      afr[k].y = hi ? 0.f     : rowp[1];
    } else {
      afr[k] = *(const v2f*)(rowp + k * 4 + hi * 2);
    }
  }

  float db[20]; int ib[20];
#pragma unroll
  for (int s = 0; s < 20; ++s) { db[s] = 3.4e38f; ib[s] = 0; }

  for (int half = 0; half < 2; ++half) {
    for (int i = lane; i < 512; i += 32) x2s[i] = x2b[half * 512 + i];
    for (int ct = 0; ct < 32; ++ct) {
      const float* colp = xb + (size_t)(half * 512 + ct * 16 + lm) * C;
      v8f acc = {};
#pragma unroll
      for (int k = 0; k < KS; ++k) {
        v2f bv;
        if (C == 3) {
          bv.x = hi ? colp[2] : colp[0];
          bv.y = hi ? 0.f     : colp[1];
        } else {
          bv = *(const v2f*)(colp + k * 4 + hi * 2);
        }
        acc = __builtin_amdgcn_wmma_f32_16x16x4_f32(
            false, afr[k], false, bv, (short)0, acc, false, false);
      }
#pragma unroll
      for (int r = 0; r < 8; ++r) {
        int mm = hi ? (r + 8) : r;
        dist[mm][ct * 16 + lm] = acc[r];
      }
    }
    __syncthreads();
    if (lane < 16) {
      float x2m = x2b[mt * 16 + lane];
      for (int cc = 0; cc < 512; ++cc) {
        int col = half * 512 + cc;
        float d = x2m + x2s[cc] - 2.f * dist[lane][cc];
        if (d < db[19]) {                    // strict <: ties keep lower index
          db[19] = d; ib[19] = col;
#pragma unroll
          for (int s = 19; s > 0; --s) {
            if (db[s] < db[s - 1]) {
              float td = db[s]; db[s] = db[s - 1]; db[s - 1] = td;
              int   ti = ib[s]; ib[s] = ib[s - 1]; ib[s - 1] = ti;
            }
          }
        }
      }
    }
    __syncthreads();
  }
  if (lane < 16) {
    size_t base = ((size_t)b * N + mt * 16 + lane) * 20;
#pragma unroll
    for (int s = 0; s < 20; ++s) idxout[base + s] = ib[s];
  }
}

// --------------------------- head edge-conv (C=3) --------------------------
// K-dim is 6 — scalar VALU is the right tool; GEMM would be padding-dominated.
__global__ void head_conv_kernel(const float* __restrict__ pts, const int* __restrict__ knn,
                                 const float* __restrict__ W /*64x6*/,
                                 const float* __restrict__ alpha, const float* __restrict__ beta,
                                 float* __restrict__ out, int B, int N) {
  int i = blockIdx.x * blockDim.x + threadIdx.x;      // b*N + n
  if (i >= B * N) return;
  int b = i / N;
  float xi0 = pts[(size_t)i * 3], xi1 = pts[(size_t)i * 3 + 1], xi2 = pts[(size_t)i * 3 + 2];
  float best[64];
#pragma unroll
  for (int o = 0; o < 64; ++o) best[o] = -3.402823e38f;
  for (int j = 0; j < 20; ++j) {
    int nb = knn[(size_t)i * 20 + j];
    const float* pj = pts + ((size_t)b * N + nb) * 3;
    float e3 = pj[0] - xi0, e4 = pj[1] - xi1, e5 = pj[2] - xi2;
#pragma unroll
    for (int o = 0; o < 64; ++o) {
      const float* wr = W + o * 6;
      float h = wr[0]*xi0 + wr[1]*xi1 + wr[2]*xi2 + wr[3]*e3 + wr[4]*e4 + wr[5]*e5;
      h = h * alpha[o] + beta[o];
      h = h >= 0.f ? h : 0.2f * h;
      best[o] = fmaxf(best[o], h);
    }
  }
#pragma unroll
  for (int o = 0; o < 64; ++o) out[(size_t)i * 64 + o] = best[o];
}

// --------------------------- WMMA edge-conv --------------------------------
// One wave owns 4 points * 20 neighbors = 80 edge rows = 5 M-tiles.
// Edge features [xi, xj-xi] built from L2-resident float4 gathers; the
// xi/diff boundary is tile-uniform (Cin % 32 == 0) so the branch is
// compile-time per unrolled kt.
template<int Cin, int Cout>
__global__ __launch_bounds__(32)
void edge_conv_wmma(const float* __restrict__ x, const int* __restrict__ knn,
                    const __bf16* __restrict__ Wb /*(Cout,2Cin)*/,
                    const float* __restrict__ alpha, const float* __restrict__ beta,
                    float* __restrict__ out, int B, int N) {
  constexpr int KT = (2 * Cin) / 32;
  constexpr int NT = Cout / 16;
  constexpr int PTS = 4;
  __shared__ float maxbuf[PTS][Cout];
  int wg = blockIdx.x;
  int b  = wg / (N / PTS);
  int n0 = (wg % (N / PTS)) * PTS;
  int lane = threadIdx.x, lm = lane & 15, hi = lane >> 4;

  for (int i = lane; i < PTS * Cout; i += 32) ((float*)maxbuf)[i] = -3.402823e38f;
  __syncthreads();

  for (int mt = 0; mt < 5; ++mt) {
    int m = mt * 16 + lm;                 // 0..79
    int p = m / 20, j = m % 20;
    int npt = n0 + p;
    int nbr = knn[((size_t)b * N + npt) * 20 + j];
    const float* xi = x + ((size_t)b * N + npt) * Cin;
    const float* xj = x + ((size_t)b * N + nbr) * Cin;

    // A fragments for all K-tiles (kept in VGPRs). Element map (16-bit A,
    // 16x32): e -> c0+e, c0 = kt*32 + hi*16.
    v16bf afr[KT];
#pragma unroll
    for (int kt = 0; kt < KT; ++kt) {
      const int c0 = kt * 32 + (hi ? 16 : 0);
      float4 q0, q1, q2, q3;
      if (kt * 32 < Cin) {                // whole tile is the xi half
        q0 = *(const float4*)(xi + c0);
        q1 = *(const float4*)(xi + c0 + 4);
        q2 = *(const float4*)(xi + c0 + 8);
        q3 = *(const float4*)(xi + c0 + 12);
      } else {                            // whole tile is the (xj - xi) half
        const int c2 = c0 - Cin;
        float4 a0 = *(const float4*)(xi + c2);
        float4 a1 = *(const float4*)(xi + c2 + 4);
        float4 a2 = *(const float4*)(xi + c2 + 8);
        float4 a3 = *(const float4*)(xi + c2 + 12);
        float4 j0 = *(const float4*)(xj + c2);
        float4 j1 = *(const float4*)(xj + c2 + 4);
        float4 j2 = *(const float4*)(xj + c2 + 8);
        float4 j3 = *(const float4*)(xj + c2 + 12);
        q0.x=j0.x-a0.x; q0.y=j0.y-a0.y; q0.z=j0.z-a0.z; q0.w=j0.w-a0.w;
        q1.x=j1.x-a1.x; q1.y=j1.y-a1.y; q1.z=j1.z-a1.z; q1.w=j1.w-a1.w;
        q2.x=j2.x-a2.x; q2.y=j2.y-a2.y; q2.z=j2.z-a2.z; q2.w=j2.w-a2.w;
        q3.x=j3.x-a3.x; q3.y=j3.y-a3.y; q3.z=j3.z-a3.z; q3.w=j3.w-a3.w;
      }
      v16bf a;
      a[0]=(__bf16)q0.x; a[1]=(__bf16)q0.y; a[2]=(__bf16)q0.z; a[3]=(__bf16)q0.w;
      a[4]=(__bf16)q1.x; a[5]=(__bf16)q1.y; a[6]=(__bf16)q1.z; a[7]=(__bf16)q1.w;
      a[8]=(__bf16)q2.x; a[9]=(__bf16)q2.y; a[10]=(__bf16)q2.z; a[11]=(__bf16)q2.w;
      a[12]=(__bf16)q3.x; a[13]=(__bf16)q3.y; a[14]=(__bf16)q3.z; a[15]=(__bf16)q3.w;
      afr[kt] = a;
    }

    for (int nt = 0; nt < NT; ++nt) {
      const int o = nt * 16 + lm;
      const __bf16* wrow = Wb + (size_t)o * (2 * Cin) + (hi ? 16 : 0);
      v8f acc = {};
#pragma unroll
      for (int kt = 0; kt < KT; ++kt) {
        // B fragment (32x16 layout): 16 contiguous bf16 = one 32B load.
        v16bf bfr = *(const v16bf*)(wrow + kt * 32);
        acc = __builtin_amdgcn_wmma_f32_16x16x32_bf16(
            false, afr[kt], false, bfr, (short)0, acc, false, false);
      }
      float al = alpha[o], bt = beta[o];
#pragma unroll
      for (int r = 0; r < 8; ++r) {
        int mm = mt * 16 + (hi ? r + 8 : r);
        int pp = mm / 20;
        float h = acc[r] * al + bt;
        h = h >= 0.f ? h : 0.2f * h;
        atomicMaxF32Shared(&maxbuf[pp][o], h);
      }
    }
  }
  __syncthreads();
  for (int i = lane; i < PTS * Cout; i += 32) {
    int pp = i / Cout, o = i % Cout;
    out[((size_t)b * N + n0 + pp) * Cout + o] = maxbuf[pp][o];
  }
}

// ------------------------------- fusion ------------------------------------
// h = fcat(BN,512) @ Wf^T(512,1024), BN+lrelu, write transposed (B,1024,N).
__global__ __launch_bounds__(32)
void fusion_wmma(const __bf16* __restrict__ fcat, const __bf16* __restrict__ Wf,
                 const float* __restrict__ alpha, const float* __restrict__ beta,
                 float* __restrict__ out, int B, int N) {
  int wg = blockIdx.x;
  int nt = wg % 64; wg /= 64;
  int mt = wg % (N / 16);
  int b  = wg / (N / 16);
  int lane = threadIdx.x, lm = lane & 15, hi = lane >> 4;
  int npt = mt * 16 + lm;
  const __bf16* arow = fcat + ((size_t)b * N + npt) * 512 + (hi ? 16 : 0);
  int o = nt * 16 + lm;
  const __bf16* brow = Wf + (size_t)o * 512 + (hi ? 16 : 0);
  v8f acc = {};
#pragma unroll
  for (int kt = 0; kt < 16; ++kt) {
    // A: two contiguous 16B runs; B: one contiguous 32B run.
    v8bf r0 = *(const v8bf*)(arow + kt * 32);
    v8bf r1 = *(const v8bf*)(arow + kt * 32 + 8);
    v16bf a;
#pragma unroll
    for (int e = 0; e < 8; ++e) { a[e] = r0[e]; a[8 + e] = r1[e]; }
    v16bf bv = *(const v16bf*)(brow + kt * 32);
    acc = __builtin_amdgcn_wmma_f32_16x16x32_bf16(
        false, a, false, bv, (short)0, acc, false, false);
  }
  float al = alpha[o], bt = beta[o];
#pragma unroll
  for (int r = 0; r < 8; ++r) {
    int mrow = mt * 16 + (hi ? r + 8 : r);
    float h = acc[r] * al + bt;
    h = h >= 0.f ? h : 0.2f * h;
    out[((size_t)b * 1024 + o) * N + mrow] = h;       // (B, 1024, N)
  }
}

// ------------------------------- launcher ----------------------------------
// Input flattening assumption (setup_inputs insertion order, recursed):
//  0:pts | head 1:W 2:b 3:g 4:be 5:m 6:v | block0 7..12 | block1 13..18 |
//  block2 19..24 | fusion 25:W 26:g 27:be 28:m 29:v 30:b
extern "C" void kernel_launch(void* const* d_in, const int* in_sizes, int n_in,
                              void* d_out, int out_size, void* d_ws, size_t ws_size,
                              hipStream_t stream) {
  const int B = 16, N = 1024, BN = B * N;
  const float* pts = (const float*)d_in[0];
  const float* HW = (const float*)d_in[1];
  const float *Hb = (const float*)d_in[2], *Hg = (const float*)d_in[3],
              *Hbe = (const float*)d_in[4], *Hm = (const float*)d_in[5],
              *Hv = (const float*)d_in[6];
  const float* W0 = (const float*)d_in[7];
  const float *b0 = (const float*)d_in[8], *g0 = (const float*)d_in[9],
              *be0 = (const float*)d_in[10], *m0 = (const float*)d_in[11],
              *v0 = (const float*)d_in[12];
  const float* W1 = (const float*)d_in[13];
  const float *b1 = (const float*)d_in[14], *g1 = (const float*)d_in[15],
              *be1 = (const float*)d_in[16], *m1 = (const float*)d_in[17],
              *v1 = (const float*)d_in[18];
  const float* W2 = (const float*)d_in[19];
  const float *b2 = (const float*)d_in[20], *g2 = (const float*)d_in[21],
              *be2 = (const float*)d_in[22], *m2 = (const float*)d_in[23],
              *v2 = (const float*)d_in[24];
  const float* WF = (const float*)d_in[25];
  const float *gF = (const float*)d_in[26], *beF = (const float*)d_in[27],
              *mF = (const float*)d_in[28], *vF = (const float*)d_in[29],
              *bF = (const float*)d_in[30];

  char* w = (char*)d_ws;
  auto alloc = [&](size_t bytes) -> char* {
    char* p = w; w += (bytes + 255) & ~(size_t)255; return p;
  };
  float* x2    = (float*)alloc((size_t)BN * 4);
  int*   idx   = (int*)  alloc((size_t)BN * 20 * 4);
  float* f0    = (float*)alloc((size_t)BN * 64 * 4);
  float* f1    = (float*)alloc((size_t)BN * 64 * 4);
  float* f2    = (float*)alloc((size_t)BN * 128 * 4);
  float* f3    = (float*)alloc((size_t)BN * 256 * 4);
  __bf16* fcat = (__bf16*)alloc((size_t)BN * 512 * 2);
  __bf16* W0b  = (__bf16*)alloc(64 * 128 * 2);
  __bf16* W1b  = (__bf16*)alloc(128 * 128 * 2);
  __bf16* W2b  = (__bf16*)alloc(256 * 256 * 2);
  __bf16* WFb  = (__bf16*)alloc(1024 * 512 * 2);
  float* aH = (float*)alloc(64 * 4);   float* bH = (float*)alloc(64 * 4);
  float* a0 = (float*)alloc(64 * 4);   float* btt0 = (float*)alloc(64 * 4);
  float* a1 = (float*)alloc(128 * 4);  float* btt1 = (float*)alloc(128 * 4);
  float* a2 = (float*)alloc(256 * 4);  float* btt2 = (float*)alloc(256 * 4);
  float* aF = (float*)alloc(1024 * 4); float* btF = (float*)alloc(1024 * 4);

  // BN-fold constants + bf16 weights
  alpha_beta_kernel<<<1, 64, 0, stream>>>(Hg, Hbe, Hm, Hv, Hb, aH, bH, 64);
  alpha_beta_kernel<<<1, 64, 0, stream>>>(g0, be0, m0, v0, b0, a0, btt0, 64);
  alpha_beta_kernel<<<2, 64, 0, stream>>>(g1, be1, m1, v1, b1, a1, btt1, 128);
  alpha_beta_kernel<<<4, 64, 0, stream>>>(g2, be2, m2, v2, b2, a2, btt2, 256);
  alpha_beta_kernel<<<16, 64, 0, stream>>>(gF, beF, mF, vF, bF, aF, btF, 1024);
  f32_to_bf16_kernel<<<(64*128 + 255) / 256, 256, 0, stream>>>(W0, W0b, 64*128);
  f32_to_bf16_kernel<<<(128*128 + 255) / 256, 256, 0, stream>>>(W1, W1b, 128*128);
  f32_to_bf16_kernel<<<(256*256 + 255) / 256, 256, 0, stream>>>(W2, W2b, 256*256);
  f32_to_bf16_kernel<<<(1024*512 + 255) / 256, 256, 0, stream>>>(WF, WFb, 1024*512);

  const int knnGrid = B * (N / 16);
  const int ecGrid  = B * (N / 4);

  // stage 0: kNN on points + head conv (3 -> 64)
  sqnorm_kernel<<<(BN + 255) / 256, 256, 0, stream>>>(pts, x2, BN, 3);
  knn_kernel<3><<<knnGrid, 32, 0, stream>>>(pts, x2, idx, B, N);
  head_conv_kernel<<<(BN + 255) / 256, 256, 0, stream>>>(pts, idx, HW, aH, bH, f0, B, N);

  // block0: kNN on f0 (64), conv 64 -> 64
  sqnorm_kernel<<<(BN + 255) / 256, 256, 0, stream>>>(f0, x2, BN, 64);
  knn_kernel<64><<<knnGrid, 32, 0, stream>>>(f0, x2, idx, B, N);
  edge_conv_wmma<64, 64><<<ecGrid, 32, 0, stream>>>(f0, idx, W0b, a0, btt0, f1, B, N);

  // block1: kNN on f1 (64), conv 64 -> 128
  sqnorm_kernel<<<(BN + 255) / 256, 256, 0, stream>>>(f1, x2, BN, 64);
  knn_kernel<64><<<knnGrid, 32, 0, stream>>>(f1, x2, idx, B, N);
  edge_conv_wmma<64, 128><<<ecGrid, 32, 0, stream>>>(f1, idx, W1b, a1, btt1, f2, B, N);

  // block2: kNN on f2 (128), conv 128 -> 256
  sqnorm_kernel<<<(BN + 255) / 256, 256, 0, stream>>>(f2, x2, BN, 128);
  knn_kernel<128><<<knnGrid, 32, 0, stream>>>(f2, x2, idx, B, N);
  edge_conv_wmma<128, 256><<<ecGrid, 32, 0, stream>>>(f2, idx, W2b, a2, btt2, f3, B, N);

  // fusion: concat(512) -> 1024, transpose
  concat_bf16_kernel<<<(BN * 512 + 255) / 256, 256, 0, stream>>>(f0, f1, f2, f3, fcat, BN);
  fusion_wmma<<<B * (N / 16) * 64, 32, 0, stream>>>(fcat, WFb, aF, btF, (float*)d_out, B, N);
}
